// GIN_52518860095603
// MI455X (gfx1250) — matrix operand order repaired and checked by reading, not compiled, not run on hardware
//
#include <hip/hip_runtime.h>
#include <hip/hip_bf16.h>

// ---------------------------------------------------------------------------
// GIN forward on MI455X (gfx1250, wave32).
// GEMMs via v_wmma_f32_16x16x32_bf16 (f32 accum), operands staged into LDS
// with CDNA5 async global->LDS DMA (ASYNCcnt), scatter via
// global_atomic_add_f32, BN via column-stat reductions + fused epilogues.
// ---------------------------------------------------------------------------

typedef __attribute__((ext_vector_type(8)))  float          v8f;
typedef __attribute__((ext_vector_type(8)))  unsigned short u16x8;
typedef __attribute__((ext_vector_type(4)))  unsigned short u16x4;
typedef __attribute__((ext_vector_type(8)))  __bf16         bf16x8;
typedef __attribute__((ext_vector_type(16))) __bf16         bf16x16;

#define HDIM   256
#define KTILE  64
#define LDPAD  72          // padded LDS row stride (bf16 elems) to spread banks
#define BN_EPS 1e-5f

#define LDS_A_ELEMS (32 * LDPAD)
#define LDS_B_BASE  LDS_A_ELEMS           // element offset of B region in smem

__device__ __forceinline__ unsigned short f2bf(float f) {
    unsigned u = __builtin_bit_cast(unsigned, f);
    unsigned r = u + 0x7FFFu + ((u >> 16) & 1u);   // round-to-nearest-even
    return (unsigned short)(r >> 16);
}
__device__ __forceinline__ float bf2f(unsigned short s) {
    return __builtin_bit_cast(float, (unsigned)s << 16);
}
__device__ __forceinline__ bf16x16 mk16(u16x8 lo, u16x8 hi) {
    bf16x8 a = __builtin_bit_cast(bf16x8, lo);
    bf16x8 b = __builtin_bit_cast(bf16x8, hi);
    return __builtin_shufflevector(a, b, 0,1,2,3,4,5,6,7,8,9,10,11,12,13,14,15);
}

// Async global->LDS copy of 16 bytes (CDNA5 async path, ASYNCcnt).
// The LDS destination is passed as a *pointer into the shared array* so the
// shared global escapes (ptrtoint) -> the memory-clobbering asm is assumed to
// write it and subsequent ds_loads are not folded away.  Per ISA 10.2, the
// low 32 bits of a generic pointer in the LDS aperture are the wave-relative
// LDS byte address, which is exactly what VDST of the async op expects.
__device__ __forceinline__ void async_b128(unsigned short* lptr,
                                           const unsigned short* gptr) {
    unsigned ldsByte = (unsigned)(unsigned long long)lptr;
    asm volatile("global_load_async_to_lds_b128 %0, %1, off"
                 :: "v"(ldsByte), "v"(gptr) : "memory");
}
__device__ __forceinline__ void wait_async0() {
    asm volatile("s_wait_asynccnt 0x0" ::: "memory");
}

// ---------------------------------------------------------------------------
// WMMA GEMM:  C[M x 256] = A[M x K](bf16, lda=K) * B[K x 256]
// Bt is B stored column-major: Bt[n*K + k]  (bf16)
// mode 0: store f32 into Cf.   mode 1: bias + relu -> bf16 into Cb.
// ---------------------------------------------------------------------------
__global__ __launch_bounds__(256)
void gemm_wmma_bf16(const unsigned short* __restrict__ A, int lda,
                    const unsigned short* __restrict__ Bt, int K,
                    float* __restrict__ Cf, unsigned short* __restrict__ Cb,
                    const float* __restrict__ bias, int mode, int M)
{
    __shared__ __align__(16) unsigned short smem[LDS_A_ELEMS + 256 * LDPAD];

    const int tid  = threadIdx.x;
    const int lane = tid & 31;
    const int wave = tid >> 5;

    const int m0blk = blockIdx.x * 32;
    const int mW    = (wave & 1) * 16;     // wave's M offset within block
    const int n0    = (wave >> 1) * 64;    // wave's N base (4 tiles of 16)

    v8f acc[4] = {};

    const int arow  = mW + (lane & 15);
    const int aoff0 = (lane < 16) ? 0 : 8;     // ISA A-frag K grouping
    const int boff0 = (lane < 16) ? 0 : 8;     // ISA B-frag K grouping

    // staging geometry (uniform across loop)
    const int sr  = tid >> 3;          // 0..31  : A row within slab
    const int scg = (tid & 7) * 8;     // 0..56  : A K-chunk within slab
    int gr = m0blk + sr;  if (gr >= M) gr = M - 1;   // clamp: padded rows
                                                     // feed only unstored C rows
    for (int kb = 0; kb < K; kb += KTILE) {
        __syncthreads();   // previous tile fully consumed before overwrite

        // ---- async stage A slab: 32 rows x 64 K (one b128 per thread) ----
        async_b128(&smem[sr * LDPAD + scg],
                   A + (long)gr * lda + kb + scg);
        // prefetch next A slab toward the device caches while this computes
        if (kb + KTILE < K)
            __builtin_prefetch(A + (long)gr * lda + kb + KTILE + scg, 0, 2);

        // ---- async stage B slab: col = tid, 64 K (8 x b128 per thread) ----
        {
            const unsigned short* src = Bt + (long)tid * K + kb;
            unsigned short*       dstp = &smem[LDS_B_BASE + tid * LDPAD];
            #pragma unroll
            for (int j = 0; j < KTILE; j += 8)
                async_b128(dstp + j, src + j);
        }
        wait_async0();
        __syncthreads();

        // ---- compute: 2 K-steps of 32, 4 N-tiles each ----
        #pragma unroll
        for (int kk = 0; kk < KTILE; kk += 32) {
            // batch all fragment loads so dscnt waits overlap the WMMAs
            u16x8 alo = *(const u16x8*)(&smem[arow * LDPAD + kk + aoff0]);
            u16x8 ahi = *(const u16x8*)(&smem[arow * LDPAD + kk + aoff0 + 16]);
            u16x8 b0[4], b1[4];
            #pragma unroll
            for (int t = 0; t < 4; ++t) {
                int col  = n0 + t * 16 + (lane & 15);
                int koff = kk + boff0;
                b0[t] = *(const u16x8*)(&smem[LDS_B_BASE + col * LDPAD + koff]);
                b1[t] = *(const u16x8*)(&smem[LDS_B_BASE + col * LDPAD + koff + 16]);
            }
            bf16x16 afrag = mk16(alo, ahi);
            #pragma unroll
            for (int t = 0; t < 4; ++t) {
                bf16x16 bfrag = mk16(b0[t], b1[t]);
                acc[t] = __builtin_amdgcn_wmma_f32_16x16x32_bf16(
                    false, afrag, false, bfrag, (short)0, acc[t], false, false);
            }
        }
    }

    // ---- writeout (ISA C layout: vgpr r -> M=r (+8 for upper half-lane)) ----
    const int coll = lane & 15;
    const int radd = (lane < 16) ? 0 : 8;
    #pragma unroll
    for (int t = 0; t < 4; ++t) {
        int col = n0 + t * 16 + coll;
        #pragma unroll
        for (int r = 0; r < 8; ++r) {
            int row = m0blk + mW + radd + r;
            if (row < M) {
                float v = acc[t][r];
                if (mode == 1) {
                    v += bias[col];
                    v = v > 0.f ? v : 0.f;
                    Cb[(long)row * HDIM + col] = f2bf(v);
                } else {
                    Cf[(long)row * HDIM + col] = v;
                }
            }
        }
    }
}

// ---------------------------------------------------------------------------
// transpose + convert f32 -> bf16:  dst[n*K + k] = src[k*Ncols + n]
// ---------------------------------------------------------------------------
__global__ void transpose_cvt(const float* __restrict__ src,
                              unsigned short* __restrict__ dst,
                              int K, int Ncols)
{
    long idx = (long)blockIdx.x * blockDim.x + threadIdx.x;
    long total = (long)K * Ncols;
    if (idx < total) {
        int k = (int)(idx / Ncols);
        int n = (int)(idx % Ncols);
        dst[(long)n * K + k] = f2bf(src[idx]);
    }
}

// ---------------------------------------------------------------------------
// h = emb[state]; also write bf16 copy into hcat[:, 0:H]
// ---------------------------------------------------------------------------
__global__ void embed_kernel(const long long* __restrict__ state,
                             const float* __restrict__ emb,
                             float* __restrict__ h,
                             unsigned short* __restrict__ hcat, int N)
{
    long idx = (long)blockIdx.x * blockDim.x + threadIdx.x;
    long total = (long)N * 64;
    if (idx >= total) return;
    int n  = (int)(idx >> 6);
    int c4 = (int)(idx & 63) << 2;
    long s = state[n];
    float4 v = *(const float4*)(emb + s * HDIM + c4);
    *(float4*)(h + (long)n * HDIM + c4) = v;
    u16x4 b = { f2bf(v.x), f2bf(v.y), f2bf(v.z), f2bf(v.w) };
    *(u16x4*)(hcat + (long)n * (5 * HDIM) + c4) = b;
}

// film = c @ film_W + film_b   (gb[0:256]=gamma, gb[256:512]=beta)
__global__ void film_kernel(const float* __restrict__ cvec,
                            const float* __restrict__ fW,
                            const float* __restrict__ fb,
                            float* __restrict__ gb)
{
    int j = blockIdx.x * blockDim.x + threadIdx.x;
    if (j < 512) {
        float s = fb[j];
        #pragma unroll
        for (int k = 0; k < 16; ++k) s += cvec[k] * fW[k * 512 + j];
        gb[j] = s;
    }
}

// hf = gamma*h + beta ; agg = 0
__global__ void apply_film(const float* __restrict__ h,
                           const float* __restrict__ gb,
                           float* __restrict__ hf,
                           float* __restrict__ agg, int N)
{
    long idx = (long)blockIdx.x * blockDim.x + threadIdx.x;
    long total = (long)N * 64;
    if (idx >= total) return;
    int c4 = (int)(idx & 63) << 2;
    long off = (idx >> 6) * HDIM + c4;
    float4 hv = *(const float4*)(h + off);
    float4 g  = *(const float4*)(gb + c4);
    float4 b  = *(const float4*)(gb + 256 + c4);
    float4 r  = { g.x*hv.x + b.x, g.y*hv.y + b.y, g.z*hv.z + b.z, g.w*hv.w + b.w };
    *(float4*)(hf + off) = r;
    float4 z = {0.f, 0.f, 0.f, 0.f};
    *(float4*)(agg + off) = z;
}

// agg[dst] += hf[src]  (segment_sum)
__global__ void scatter_add(const float* __restrict__ hf,
                            const int* __restrict__ src,
                            const int* __restrict__ dst,
                            float* __restrict__ agg, long total)
{
    long stride = (long)gridDim.x * blockDim.x;
    for (long i = (long)blockIdx.x * blockDim.x + threadIdx.x; i < total; i += stride) {
        long e  = i >> 6;
        int  c4 = (int)(i & 63) << 2;
        int  s  = src[e];
        int  d  = dst[e];
        float4 v = *(const float4*)(hf + (long)s * HDIM + c4);
        float* p = agg + (long)d * HDIM + c4;
        atomicAdd(p + 0, v.x);
        atomicAdd(p + 1, v.y);
        atomicAdd(p + 2, v.z);
        atomicAdd(p + 3, v.w);
    }
}

// zb = bf16(hf + agg)
__global__ void add_cvt(const float* __restrict__ hf,
                        const float* __restrict__ agg,
                        unsigned short* __restrict__ zb, int N)
{
    long idx = (long)blockIdx.x * blockDim.x + threadIdx.x;
    long total = (long)N * 64;
    if (idx >= total) return;
    long off = ((idx >> 6) * HDIM) + (((int)(idx & 63)) << 2);
    float4 a = *(const float4*)(hf + off);
    float4 b = *(const float4*)(agg + off);
    u16x4 o = { f2bf(a.x + b.x), f2bf(a.y + b.y), f2bf(a.z + b.z), f2bf(a.w + b.w) };
    *(u16x4*)(zb + off) = o;
}

__global__ void zero_stats(float* __restrict__ sum, float* __restrict__ sumsq)
{
    int t = threadIdx.x;
    if (t < 256) { sum[t] = 0.f; sumsq[t] = 0.f; }
}

// per-column sum / sumsq (atomics across row-chunks)
__global__ void col_stats(const float* __restrict__ X,
                          float* __restrict__ sum, float* __restrict__ sumsq,
                          int M, int rowsPerBlock)
{
    int col = threadIdx.x;                 // 256 threads = 256 cols
    int r0  = blockIdx.x * rowsPerBlock;
    int r1  = r0 + rowsPerBlock; if (r1 > M) r1 = M;
    float s = 0.f, q = 0.f;
    for (int r = r0; r < r1; ++r) {
        float x = X[(long)r * HDIM + col];
        s += x; q += x * x;
    }
    atomicAdd(&sum[col], s);
    atomicAdd(&sumsq[col], q);
}

// out_bf16 = bf16(relu(bn(X)))
__global__ void bnrelu_b16(const float* __restrict__ X,
                           const float* __restrict__ sum,
                           const float* __restrict__ sumsq,
                           const float* __restrict__ g,
                           const float* __restrict__ b,
                           unsigned short* __restrict__ out, int N)
{
    long idx = (long)blockIdx.x * blockDim.x + threadIdx.x;
    long total = (long)N * 64;
    if (idx >= total) return;
    int c4 = (int)(idx & 63) << 2;
    long off = (idx >> 6) * HDIM + c4;
    float invN = 1.f / (float)N;
    float4 x  = *(const float4*)(X + off);
    float4 sm = *(const float4*)(sum + c4);
    float4 sq = *(const float4*)(sumsq + c4);
    float4 gg = *(const float4*)(g + c4);
    float4 bb = *(const float4*)(b + c4);
    float m0 = sm.x*invN, m1 = sm.y*invN, m2 = sm.z*invN, m3 = sm.w*invN;
    float y0 = gg.x*(x.x-m0)*rsqrtf(sq.x*invN - m0*m0 + BN_EPS) + bb.x;
    float y1 = gg.y*(x.y-m1)*rsqrtf(sq.y*invN - m1*m1 + BN_EPS) + bb.y;
    float y2 = gg.z*(x.z-m2)*rsqrtf(sq.z*invN - m2*m2 + BN_EPS) + bb.z;
    float y3 = gg.w*(x.w-m3)*rsqrtf(sq.w*invN - m3*m3 + BN_EPS) + bb.w;
    y0 = y0 > 0.f ? y0 : 0.f;  y1 = y1 > 0.f ? y1 : 0.f;
    y2 = y2 > 0.f ? y2 : 0.f;  y3 = y3 > 0.f ? y3 : 0.f;
    u16x4 o = { f2bf(y0), f2bf(y1), f2bf(y2), f2bf(y3) };
    *(u16x4*)(out + off) = o;
}

// h = relu(bn(X)) (f32)  and  hcat[:, layerOff:layerOff+H] = bf16(same)
__global__ void bnrelu_out(const float* __restrict__ X,
                           const float* __restrict__ sum,
                           const float* __restrict__ sumsq,
                           const float* __restrict__ g,
                           const float* __restrict__ b,
                           float* __restrict__ h,
                           unsigned short* __restrict__ hcat,
                           int layerOff, int N)
{
    long idx = (long)blockIdx.x * blockDim.x + threadIdx.x;
    long total = (long)N * 64;
    if (idx >= total) return;
    int n  = (int)(idx >> 6);
    int c4 = (int)(idx & 63) << 2;
    long off = (long)n * HDIM + c4;
    float invN = 1.f / (float)N;
    float4 x  = *(const float4*)(X + off);
    float4 sm = *(const float4*)(sum + c4);
    float4 sq = *(const float4*)(sumsq + c4);
    float4 gg = *(const float4*)(g + c4);
    float4 bb = *(const float4*)(b + c4);
    float m0 = sm.x*invN, m1 = sm.y*invN, m2 = sm.z*invN, m3 = sm.w*invN;
    float y0 = gg.x*(x.x-m0)*rsqrtf(sq.x*invN - m0*m0 + BN_EPS) + bb.x;
    float y1 = gg.y*(x.y-m1)*rsqrtf(sq.y*invN - m1*m1 + BN_EPS) + bb.y;
    float y2 = gg.z*(x.z-m2)*rsqrtf(sq.z*invN - m2*m2 + BN_EPS) + bb.z;
    float y3 = gg.w*(x.w-m3)*rsqrtf(sq.w*invN - m3*m3 + BN_EPS) + bb.w;
    y0 = y0 > 0.f ? y0 : 0.f;  y1 = y1 > 0.f ? y1 : 0.f;
    y2 = y2 > 0.f ? y2 : 0.f;  y3 = y3 > 0.f ? y3 : 0.f;
    float4 yo = { y0, y1, y2, y3 };
    *(float4*)(h + off) = yo;
    u16x4 o = { f2bf(y0), f2bf(y1), f2bf(y2), f2bf(y3) };
    *(u16x4*)(hcat + (long)n * (5 * HDIM) + layerOff + c4) = o;
}

// out[n,:] = r1[n,:] @ ro_W2 + ro_b2   (256 x 2)
__global__ void final_out(const unsigned short* __restrict__ r1,
                          const float* __restrict__ W2,
                          const float* __restrict__ b2,
                          float* __restrict__ out, int N)
{
    int n = blockIdx.x * blockDim.x + threadIdx.x;
    if (n >= N) return;
    float a0 = b2[0], a1 = b2[1];
    const unsigned short* row = r1 + (long)n * HDIM;
    #pragma unroll 8
    for (int k = 0; k < HDIM; ++k) {
        float x = bf2f(row[k]);
        a0 += x * W2[k * 2 + 0];
        a1 += x * W2[k * 2 + 1];
    }
    out[(long)n * 2 + 0] = a0;
    out[(long)n * 2 + 1] = a1;
}

// ---------------------------------------------------------------------------
extern "C" void kernel_launch(void* const* d_in, const int* in_sizes, int n_in,
                              void* d_out, int out_size, void* d_ws, size_t ws_size,
                              hipStream_t stream)
{
    const int N = in_sizes[0];
    const int E = in_sizes[1];
    const int H = HDIM, L = 4;

    const long long* state = (const long long*)d_in[0];
    const int*   src    = (const int*)  d_in[1];
    const int*   dst    = (const int*)  d_in[2];
    const float* cvec   = (const float*)d_in[3];
    const float* emb    = (const float*)d_in[4];
    const float* film_W = (const float*)d_in[5];
    const float* film_b = (const float*)d_in[6];
    const float* W1     = (const float*)d_in[7];
    const float* bn1_g  = (const float*)d_in[8];
    const float* bn1_b  = (const float*)d_in[9];
    const float* W2     = (const float*)d_in[10];
    const float* bn2_g  = (const float*)d_in[11];
    const float* bn2_b  = (const float*)d_in[12];
    const float* roW1   = (const float*)d_in[13];
    const float* rob1   = (const float*)d_in[14];
    const float* roW2   = (const float*)d_in[15];
    const float* rob2   = (const float*)d_in[16];
    float* out = (float*)d_out;

    // ---- workspace layout (all sizes multiple of 256B) ----
    char* base = (char*)d_ws;
    size_t szNH_f32  = (size_t)N * H * 4;
    size_t szNH_b16  = (size_t)N * H * 2;
    size_t szHcat    = (size_t)N * 5 * H * 2;
    float*          h     = (float*)base;                       base += szNH_f32;
    float*          hf    = (float*)base;                       base += szNH_f32;  // also t1
    float*          agg   = (float*)base;                       base += szNH_f32;  // also t2
    unsigned short* zb    = (unsigned short*)base;              base += szNH_b16;  // also t1b, r1b
    unsigned short* hcat  = (unsigned short*)base;              base += szHcat;
    unsigned short* W1t   = (unsigned short*)base;              base += (size_t)L * H * H * 2;
    unsigned short* W2t   = (unsigned short*)base;              base += (size_t)L * H * H * 2;
    unsigned short* roW1t = (unsigned short*)base;              base += (size_t)5 * H * H * 2;
    float*          gb    = (float*)base;                       base += 512 * 4;
    float*          csum  = (float*)base;                       base += 256 * 4;
    float*          csq   = (float*)base;                       base += 256 * 4;
    (void)ws_size; (void)n_in; (void)out_size;

    float* t1  = hf;
    float* t2  = agg;
    unsigned short* t1b = zb;
    unsigned short* r1b = zb;

    const int TB = 256;
    const long nh4    = (long)N * 64;              // N*H/4 vec4 work-items
    const int  gNH4   = (int)((nh4 + TB - 1) / TB);
    const int  gGemm  = (N + 31) / 32;
    const int  gStats = (N + 127) / 128;
    const long eTot   = (long)E * 64;

    // ---- weight prep: transpose + bf16 convert ----
    for (int l = 0; l < L; ++l) {
        transpose_cvt<<<(H*H + TB - 1)/TB, TB, 0, stream>>>(W1 + (size_t)l*H*H, W1t + (size_t)l*H*H, H, H);
        transpose_cvt<<<(H*H + TB - 1)/TB, TB, 0, stream>>>(W2 + (size_t)l*H*H, W2t + (size_t)l*H*H, H, H);
    }
    transpose_cvt<<<(5*H*H + TB - 1)/TB, TB, 0, stream>>>(roW1, roW1t, 5*H, H);

    // ---- embedding gather ----
    embed_kernel<<<gNH4, TB, 0, stream>>>(state, emb, h, hcat, N);

    // ---- GIN layers ----
    for (int l = 0; l < L; ++l) {
        film_kernel<<<2, 256, 0, stream>>>(cvec, film_W + (size_t)l*16*512, film_b + (size_t)l*512, gb);
        apply_film<<<gNH4, TB, 0, stream>>>(h, gb, hf, agg, N);
        scatter_add<<<16384, TB, 0, stream>>>(hf, src, dst, agg, eTot);
        add_cvt<<<gNH4, TB, 0, stream>>>(hf, agg, zb, N);

        gemm_wmma_bf16<<<gGemm, TB, 0, stream>>>(zb, H, W1t + (size_t)l*H*H, H, t1, nullptr, nullptr, 0, N);
        zero_stats<<<1, 256, 0, stream>>>(csum, csq);
        col_stats<<<gStats, 256, 0, stream>>>(t1, csum, csq, N, 128);
        bnrelu_b16<<<gNH4, TB, 0, stream>>>(t1, csum, csq, bn1_g + l*H, bn1_b + l*H, t1b, N);

        gemm_wmma_bf16<<<gGemm, TB, 0, stream>>>(t1b, H, W2t + (size_t)l*H*H, H, t2, nullptr, nullptr, 0, N);
        zero_stats<<<1, 256, 0, stream>>>(csum, csq);
        col_stats<<<gStats, 256, 0, stream>>>(t2, csum, csq, N, 128);
        bnrelu_out<<<gNH4, TB, 0, stream>>>(t2, csum, csq, bn2_g + l*H, bn2_b + l*H, h, hcat, (l+1)*H, N);
    }

    // ---- readout ----
    gemm_wmma_bf16<<<gGemm, TB, 0, stream>>>(hcat, 5*H, roW1t, 5*H, nullptr, r1b, rob1, 1, N);
    final_out<<<(N + TB - 1)/TB, TB, 0, stream>>>(r1b, roW2, rob2, out, N);
}
